// SoftMoELayerWrapper_52192442581479
// MI455X (gfx1250) — matrix operand order repaired and checked by reading, not compiled, not run on hardware
//
#include <hip/hip_runtime.h>
#include <math.h>

// ---------------------------------------------------------------------------
// Problem constants (match reference)
// ---------------------------------------------------------------------------
constexpr int cB = 8, cM = 1024, cD = 768;
constexpr int cN = 32, cP = 32, cK = 8, cH = 2048;
constexpr int cNP = cN * cP;          // 1024
constexpr int cT  = cB * cM;          // 8192 tokens
constexpr int TS  = 32;               // ACMoE token chunks
constexpr int TC  = cT / TS;          // tokens per chunk

typedef __attribute__((ext_vector_type(16))) __bf16 v16bf;
typedef __attribute__((ext_vector_type(8)))  float  v8f;
typedef __attribute__((ext_vector_type(8)))  unsigned short us8;
typedef __attribute__((ext_vector_type(4)))  unsigned short us4;
typedef __attribute__((ext_vector_type(4)))  float  f4;

__device__ __forceinline__ unsigned short f2bf(float f) {
  union { float f; unsigned int u; } x; x.f = f;
  unsigned int r = x.u + 0x7FFFu + ((x.u >> 16) & 1u);   // RNE
  return (unsigned short)(r >> 16);
}

// ---------------------------------------------------------------------------
// Normalize tokens: xn = x / max(||x||, eps)   (one block per token)
// ---------------------------------------------------------------------------
__global__ __launch_bounds__(256) void normalize_x_kernel(
    const float* __restrict__ x, float* __restrict__ xn) {
  __shared__ float sh[256];
  const long t = blockIdx.x;
  const int tid = threadIdx.x;
  float s = 0.f;
  for (int d = tid; d < cD; d += 256) { float v = x[t * cD + d]; s += v * v; }
  sh[tid] = s; __syncthreads();
  for (int st = 128; st > 0; st >>= 1) { if (tid < st) sh[tid] += sh[tid + st]; __syncthreads(); }
  const float inv = 1.f / fmaxf(sqrtf(sh[0]), 1e-12f);
  for (int d = tid; d < cD; d += 256) xn[t * cD + d] = x[t * cD + d] * inv;
}

// ---------------------------------------------------------------------------
// Normalize phi columns (norm over D), scale, emit bf16 [D][N*P]
// ---------------------------------------------------------------------------
__global__ __launch_bounds__(256) void normalize_phi_kernel(
    const float* __restrict__ phi, const float* __restrict__ scale,
    unsigned short* __restrict__ phin_bf) {
  __shared__ float sh[256];
  const int c = blockIdx.x;
  const int tid = threadIdx.x;
  float s = 0.f;
  for (int d = tid; d < cD; d += 256) { float v = phi[(long)d * cNP + c]; s += v * v; }
  sh[tid] = s; __syncthreads();
  for (int st = 128; st > 0; st >>= 1) { if (tid < st) sh[tid] += sh[tid + st]; __syncthreads(); }
  const float inv = scale[0] / fmaxf(sqrtf(sh[0]), 1e-12f);
  for (int d = tid; d < cD; d += 256)
    phin_bf[(long)d * cNP + c] = f2bf(phi[(long)d * cNP + c] * inv);
}

// ---------------------------------------------------------------------------
// ACMoE: per-(chunk,cluster) partial sums. Grid = (D/256, TS).
// pass==0: sum of xn ; pass==1: sum of |xn - mean|
// ---------------------------------------------------------------------------
__global__ __launch_bounds__(256) void acmoe_seg_kernel(
    const float* __restrict__ xn, const int* __restrict__ assign,
    const float* __restrict__ mean, float* __restrict__ partial, int pass) {
  __shared__ float ls[cN][256];
  const int tid = threadIdx.x;
  const int d = blockIdx.x * 256 + tid;
  const int ts = blockIdx.y;
  for (int c = 0; c < cN; ++c) ls[c][tid] = 0.f;
  for (int t = ts * TC; t < (ts + 1) * TC; ++t) {
    const int c = assign[(long)t * cK];
    float v = xn[(long)t * cD + d];
    if (pass) v = fabsf(v - mean[(long)c * cD + d]);
    ls[c][tid] += v;
  }
  for (int c = 0; c < cN; ++c)
    partial[((long)ts * cN + c) * cD + d] = ls[c][tid];
}

__global__ __launch_bounds__(256) void acmoe_reduce_kernel(
    const float* __restrict__ partial, float* __restrict__ out) {
  const long i = (long)blockIdx.x * 256 + threadIdx.x;     // < N*D
  float s = 0.f;
#pragma unroll
  for (int ts = 0; ts < TS; ++ts) s += partial[(long)ts * cN * cD + i];
  out[i] = s;
}

__global__ void acmoe_zero_cnt_kernel(int* __restrict__ icnt) {
  if (threadIdx.x < cN) icnt[threadIdx.x] = 0;
}
// integer atomic adds are associative -> deterministic counts
__global__ __launch_bounds__(256) void acmoe_count_kernel(
    const int* __restrict__ assign, int* __restrict__ icnt) {
  const long t = (long)blockIdx.x * 256 + threadIdx.x;
  atomicAdd(&icnt[assign[t * cK]], 1);
}

__global__ __launch_bounds__(256) void acmoe_mean_kernel(
    const float* __restrict__ sum, const int* __restrict__ icnt,
    float* __restrict__ mean) {
  const long i = (long)blockIdx.x * 256 + threadIdx.x;     // < N*D
  const float c = (float)icnt[i / cD];
  mean[i] = (c > 0.f) ? sum[i] / fmaxf(c, 1.f) : 0.f;
}

// W = 1/(mad+0.35); clamp at 5*mean(W); W /= mean(W_clamped). One block.
__global__ __launch_bounds__(256) void acmoe_finalize_kernel(
    const float* __restrict__ madsum, const int* __restrict__ icnt,
    float* __restrict__ W) {
  __shared__ float sh[256];
  const int tid = threadIdx.x;
  const int ND = cN * cD;
  float part = 0.f;
  for (int i = tid; i < ND; i += 256) {
    const float c = (float)icnt[i / cD];
    const float wm = (c > 0.f) ? madsum[i] / fmaxf(c, 1.f) : 0.f;
    const float w = 1.f / (wm + 0.35f);
    W[i] = w; part += w;
  }
  sh[tid] = part; __syncthreads();
  for (int st = 128; st > 0; st >>= 1) { if (tid < st) sh[tid] += sh[tid + st]; __syncthreads(); }
  const float clampv = 5.f * (sh[0] / (float)ND);
  __syncthreads();
  part = 0.f;
  for (int i = tid; i < ND; i += 256) {
    const float w = fminf(W[i], clampv);
    W[i] = w; part += w;
  }
  sh[tid] = part; __syncthreads();
  for (int st = 128; st > 0; st >>= 1) { if (tid < st) sh[tid] += sh[tid + st]; __syncthreads(); }
  const float invmean = (float)ND / sh[0];
  __syncthreads();
  for (int i = tid; i < ND; i += 256) W[i] *= invmean;
}

__global__ __launch_bounds__(256) void acmoe_apply_kernel(
    const float* __restrict__ xn, const int* __restrict__ assign,
    const float* __restrict__ W, unsigned short* __restrict__ xn_bf) {
  const long t = blockIdx.x;
  const int c = assign[t * cK];
  for (int d = threadIdx.x; d < cD; d += 256)
    xn_bf[t * cD + d] = f2bf(xn[t * cD + d] * W[(long)c * cD + d]);
}

// ---------------------------------------------------------------------------
// WMMA bf16 GEMM:  C[z] = A[z] (MxK bf16, row-major) * B[z%bMod] (KxN, rowmaj)
// 256 threads = 8 waves, one 16x16 tile each; block tile BM x BN; K-step 32.
// A staged via global_load_async_to_lds; B vector-loaded + transposed in LDS.
// EPI: 0 = f32 store | 1 = bf16 | 2 = +bias, exact GELU, bf16 | 3 = +bias, bf16
// ---------------------------------------------------------------------------
template <int BM, int BN, bool BBF, int EPI>
__global__ __launch_bounds__(256) void wmma_gemm_kernel(
    const unsigned short* __restrict__ Ah, const void* __restrict__ Bp,
    float* __restrict__ Cf, unsigned short* __restrict__ Ch,
    const float* __restrict__ bias,
    int Nc, int Kd, long aStride, long bStride, int bMod,
    long cStride, int biasMod) {
  constexpr int WTN = BN / 16;
  constexpr int EA  = BM * 32 / 256;    // A elems/thread: 8 (BM=64) or 4 (BM=32)
  constexpr int EB  = BN * 32 / 256;    // B elems/thread: 4 (BN=32) or 8 (BN=64)
  __shared__ unsigned short lA[BM][40];   // stride 80B: 16B-aligned wide ops
  __shared__ unsigned short lBt[BN][40];  // transposed: [col][k]

  const int tid = threadIdx.x, lane = tid & 31, wave = tid >> 5;
  const int wr = (wave / WTN) * 16, wc = (wave % WTN) * 16;
  const int z = blockIdx.z;
  const int rowBase = blockIdx.y * BM, colBase = blockIdx.x * BN;

  const unsigned short* A = Ah + (long)z * aStride;
  const int zb = z % bMod;
  const float*          Bf = (const float*)Bp + (long)zb * bStride;
  const unsigned short* Bh = (const unsigned short*)Bp + (long)zb * bStride;

  // --- per-thread staging coordinates (constant; K advance is += 32 elems) ---
  const int aRow = (tid * EA) >> 5;
  const int aK   = (tid * EA) & 31;
  const unsigned short* aPtr = A + (long)(rowBase + aRow) * Kd + aK;
  const unsigned ldsA = (unsigned)(unsigned long long)&lA[aRow][aK];

  const int bK = tid / (BN / EB);                 // k row within tile (0..31)
  const int bC = (tid % (BN / EB)) * EB;          // first col handled
  const long bOff0 = (long)bK * Nc + (colBase + bC);

  v8f acc = {0.f, 0.f, 0.f, 0.f, 0.f, 0.f, 0.f, 0.f};
  union Frag { v16bf v; unsigned int u[8]; };

  for (int k0 = 0; k0 < Kd; k0 += 32) {
    // ---- B tile -> registers (overlaps previous tile's WMMA) ----
    unsigned short bTmp[EB];
    if (BBF) {
      if (EB == 8) {
        us8 v = *(const us8*)(Bh + bOff0 + (long)k0 * Nc);
#pragma unroll
        for (int j = 0; j < EB; ++j) bTmp[j] = v[j];
      } else {
        us4 v = *(const us4*)(Bh + bOff0 + (long)k0 * Nc);
#pragma unroll
        for (int j = 0; j < EB; ++j) bTmp[j] = v[j];
      }
    } else {
      const float* p = Bf + bOff0 + (long)k0 * Nc;
      f4 v0 = *(const f4*)p;
#pragma unroll
      for (int j = 0; j < (EB < 4 ? EB : 4); ++j) bTmp[j] = f2bf(v0[j]);
      if (EB == 8) {
        f4 v1 = *(const f4*)(p + 4);
#pragma unroll
        for (int j = 0; j < 4; ++j) bTmp[4 + j] = f2bf(v1[j]);
      }
    }

    __syncthreads();   // previous tile fully consumed

    // ---- A tile: async DMA global -> LDS (CDNA5 path, ASYNCcnt) ----
    {
      const unsigned long long ga = (unsigned long long)(aPtr + k0);
      if (EA == 8)
        asm volatile("global_load_async_to_lds_b128 %0, %1, off"
                     :: "v"(ldsA), "v"(ga) : "memory");
      else
        asm volatile("global_load_async_to_lds_b64 %0, %1, off"
                     :: "v"(ldsA), "v"(ga) : "memory");
    }

    // ---- B tile: transpose-scatter into LDS ----
#pragma unroll
    for (int j = 0; j < EB; ++j) lBt[bC + j][bK] = bTmp[j];

    asm volatile("s_wait_asynccnt 0x0" ::: "memory");
    __syncthreads();   // tile visible to all waves

    // ---- fragments (ISA 7.12.2 layouts) + WMMA ----
    Frag a, b;
    const int rl  = wr + (lane & 15);
    const int kbA = (lane < 16) ? 0 : 8;
#pragma unroll
    for (int j = 0; j < 4; ++j) {
      a.u[j]     = *(const unsigned int*)&lA[rl][kbA + 2 * j];
      a.u[j + 4] = *(const unsigned int*)&lA[rl][16 + kbA + 2 * j];
    }
    const int cl  = wc + (lane & 15);
    const int kbB = (lane < 16) ? 0 : 8;
#pragma unroll
    for (int j = 0; j < 4; ++j) {
      b.u[j]     = *(const unsigned int*)&lBt[cl][kbB + 2 * j];
      b.u[j + 4] = *(const unsigned int*)&lBt[cl][16 + kbB + 2 * j];
    }

    acc = __builtin_amdgcn_wmma_f32_16x16x32_bf16(
        false, a.v, false, b.v, (short)0, acc, false, false);
  }

  // ---- epilogue: VGPR i -> row i (lanes 0-15) / i+8 (lanes 16-31) ----
  const int col = colBase + wc + (lane & 15);
  const float bv = (EPI >= 2) ? bias[(long)(z % biasMod) * Nc + col] : 0.f;
#pragma unroll
  for (int i = 0; i < 8; ++i) {
    const int r = rowBase + wr + i + ((lane >> 4) << 3);
    float v = acc[i];
    if (EPI == 2) { v += bv; v = 0.5f * v * (1.f + erff(v * 0.70710678118654752f)); }
    if (EPI == 3) { v += bv; }
    const long ci = (long)z * cStride + (long)r * Nc + col;
    if (EPI == 0) Cf[ci] = v;
    else          Ch[ci] = f2bf(v);
  }
}

// ---------------------------------------------------------------------------
// Row softmax over N*P (comb, bf16) + deterministic top-8
// ---------------------------------------------------------------------------
__global__ __launch_bounds__(256) void row_softmax_topk_kernel(
    const float* __restrict__ logits, unsigned short* __restrict__ comb_bf,
    float* __restrict__ mix, float* __restrict__ clus) {
  __shared__ float shv[256];
  __shared__ int   shi[256];
  __shared__ int   sel[cK];
  const long row = blockIdx.x;
  const float* a = logits + row * cNP;
  const int tid = threadIdx.x;

  float m = -3.4e38f;
  for (int j = tid; j < cNP; j += 256) m = fmaxf(m, a[j]);
  shv[tid] = m; __syncthreads();
  for (int s = 128; s > 0; s >>= 1) { if (tid < s) shv[tid] = fmaxf(shv[tid], shv[tid + s]); __syncthreads(); }
  m = shv[0]; __syncthreads();

  float ssum = 0.f;
  for (int j = tid; j < cNP; j += 256) ssum += __expf(a[j] - m);
  shv[tid] = ssum; __syncthreads();
  for (int s = 128; s > 0; s >>= 1) { if (tid < s) shv[tid] += shv[tid + s]; __syncthreads(); }
  const float inv = 1.f / shv[0]; __syncthreads();

  for (int j = tid; j < cNP; j += 256)
    comb_bf[row * cNP + j] = f2bf(__expf(a[j] - m) * inv);

  for (int it = 0; it < cK; ++it) {
    float bv = -3.4e38f; int bi = 0x7FFFFFFF;
    for (int j = tid; j < cNP; j += 256) {
      bool taken = false;
      for (int q = 0; q < it; ++q) if (sel[q] == j) taken = true;
      const float v = a[j];
      if (!taken && (v > bv || (v == bv && j < bi))) { bv = v; bi = j; }
    }
    shv[tid] = bv; shi[tid] = bi; __syncthreads();
    for (int s = 128; s > 0; s >>= 1) {
      if (tid < s) {
        if (shv[tid + s] > shv[tid] ||
            (shv[tid + s] == shv[tid] && shi[tid + s] < shi[tid])) {
          shv[tid] = shv[tid + s]; shi[tid] = shi[tid + s];
        }
      }
      __syncthreads();
    }
    if (tid == 0) {
      sel[it] = shi[0];
      mix [row * cK + it] = shv[0];
      clus[row * cK + it] = (float)(shi[0] >> 5);   // slot / P
    }
    __syncthreads();
  }
}

// ---------------------------------------------------------------------------
// Column softmax over tokens m (disp), written transposed bf16 dispT[b][np][m]
// ---------------------------------------------------------------------------
__global__ __launch_bounds__(256) void col_softmax_kernel(
    const float* __restrict__ logits, unsigned short* __restrict__ dispT_bf) {
  __shared__ float sh[256];
  const int b   = blockIdx.x >> 10;
  const int col = blockIdx.x & (cNP - 1);
  const float* base = logits + (long)b * cM * cNP + col;
  const int tid = threadIdx.x;

  float m = -3.4e38f;
  for (int j = tid; j < cM; j += 256) m = fmaxf(m, base[(long)j * cNP]);
  sh[tid] = m; __syncthreads();
  for (int s = 128; s > 0; s >>= 1) { if (tid < s) sh[tid] = fmaxf(sh[tid], sh[tid + s]); __syncthreads(); }
  m = sh[0]; __syncthreads();

  float ssum = 0.f;
  for (int j = tid; j < cM; j += 256) ssum += __expf(base[(long)j * cNP] - m);
  sh[tid] = ssum; __syncthreads();
  for (int s = 128; s > 0; s >>= 1) { if (tid < s) sh[tid] += sh[tid + s]; __syncthreads(); }
  const float inv = 1.f / sh[0];

  unsigned short* o = dispT_bf + ((long)b * cNP + col) * cM;
  for (int j = tid; j < cM; j += 256)
    o[j] = f2bf(__expf(base[(long)j * cNP] - m) * inv);
}

// ---------------------------------------------------------------------------
// Launch
// ---------------------------------------------------------------------------
extern "C" void kernel_launch(void* const* d_in, const int* in_sizes, int n_in,
                              void* d_out, int out_size, void* d_ws, size_t ws_size,
                              hipStream_t stream) {
  (void)in_sizes; (void)n_in; (void)out_size; (void)ws_size;
  const float* x      = (const float*)d_in[0];
  const int*   assign = (const int*)  d_in[1];
  const float* phi    = (const float*)d_in[2];
  const float* scale  = (const float*)d_in[3];
  const float* w1     = (const float*)d_in[4];
  const float* b1     = (const float*)d_in[5];
  const float* w2     = (const float*)d_in[6];
  const float* b2     = (const float*)d_in[7];
  float* out = (float*)d_out;

  char* ws = (char*)d_ws;
  size_t o = 0;
  auto take = [&](size_t bytes) { size_t r = o; o = (o + bytes + 255) & ~(size_t)255; return r; };
  float*          xn      = (float*)         (ws + take((size_t)cT * cD * 4));
  unsigned short* xn_bf   = (unsigned short*)(ws + take((size_t)cT * cD * 2));
  unsigned short* phin_bf = (unsigned short*)(ws + take((size_t)cD * cNP * 2));
  float*          logits  = (float*)         (ws + take((size_t)cT * cNP * 4));
  unsigned short* comb_bf = (unsigned short*)(ws + take((size_t)cT * cNP * 2));
  unsigned short* dispT   = (unsigned short*)(ws + take((size_t)cB * cNP * cM * 2));
  unsigned short* xs_bf   = (unsigned short*)(ws + take((size_t)cB * cNP * cD * 2));
  unsigned short* h_bf    = (unsigned short*)(ws + take((size_t)cB * cNP * cH * 2));
  unsigned short* ys_bf   = (unsigned short*)(ws + take((size_t)cB * cNP * cD * 2));
  float*          partial = (float*)         (ws + take((size_t)TS * cN * cD * 4));
  float*          segsum  = (float*)         (ws + take((size_t)cN * cD * 4));
  float*          segmad  = (float*)         (ws + take((size_t)cN * cD * 4));
  float*          segmean = (float*)         (ws + take((size_t)cN * cD * 4));
  float*          Wfac    = (float*)         (ws + take((size_t)cN * cD * 4));
  int*            icnt    = (int*)           (ws + take((size_t)cN * 4));

  float* mix  = out + (size_t)cB * cM * cD;
  float* clus = mix + (size_t)cB * cM * cK;

  // 1) normalize
  normalize_x_kernel  <<<cT,  256, 0, stream>>>(x, xn);
  normalize_phi_kernel<<<cNP, 256, 0, stream>>>(phi, scale, phin_bf);

  // 2) ACMoE reweighting
  acmoe_zero_cnt_kernel<<<1, 32, 0, stream>>>(icnt);
  acmoe_count_kernel   <<<cT / 256, 256, 0, stream>>>(assign, icnt);
  acmoe_seg_kernel     <<<dim3(cD / 256, TS), 256, 0, stream>>>(xn, assign, segmean, partial, 0);
  acmoe_reduce_kernel  <<<(cN * cD) / 256, 256, 0, stream>>>(partial, segsum);
  acmoe_mean_kernel    <<<(cN * cD) / 256, 256, 0, stream>>>(segsum, icnt, segmean);
  acmoe_seg_kernel     <<<dim3(cD / 256, TS), 256, 0, stream>>>(xn, assign, segmean, partial, 1);
  acmoe_reduce_kernel  <<<(cN * cD) / 256, 256, 0, stream>>>(partial, segmad);
  acmoe_finalize_kernel<<<1, 256, 0, stream>>>(segmad, icnt, Wfac);
  acmoe_apply_kernel   <<<cT, 256, 0, stream>>>(xn, assign, Wfac, xn_bf);

  // 3) logits = xn (8192x768) * phin (768x1024) -> f32
  wmma_gemm_kernel<64, 32, true, 0><<<dim3(cNP / 32, cT / 64, 1), 256, 0, stream>>>(
      xn_bf, phin_bf, logits, nullptr, nullptr, cNP, cD, 0L, 0L, 1, 0L, 1);

  // 4) softmaxes + top-k
  row_softmax_topk_kernel<<<cT, 256, 0, stream>>>(logits, comb_bf, mix, clus);
  col_softmax_kernel<<<cB * cNP, 256, 0, stream>>>(logits, dispT);

  // 5) xs[b] = dispT[b] (1024x1024) * xn[b] (1024x768) -> bf16
  wmma_gemm_kernel<64, 32, true, 1><<<dim3(cD / 32, cNP / 64, cB), 256, 0, stream>>>(
      dispT, xn_bf, nullptr, xs_bf, nullptr, cD, cM,
      (long)cNP * cM, (long)cM * cD, cB, (long)cNP * cD, 1);

  // 6) h[b,n] = GELU(xs (32x768) * w1[n] (768x2048) + b1[n]) -> bf16
  wmma_gemm_kernel<32, 64, false, 2><<<dim3(cH / 64, 1, cB * cN), 256, 0, stream>>>(
      xs_bf, w1, nullptr, h_bf, b1, cH, cD,
      (long)cP * cD, (long)cD * cH, cN, (long)cP * cH, cN);

  // 7) ys[b,n] = h (32x2048) * w2[n] (2048x768) + b2[n] -> bf16
  wmma_gemm_kernel<32, 64, false, 3><<<dim3(cD / 64, 1, cB * cN), 256, 0, stream>>>(
      h_bf, w2, nullptr, ys_bf, b2, cD, cH,
      (long)cP * cH, (long)cH * cD, cN, (long)cP * cD, cN);

  // 8) y[b] = comb[b] (1024x1024) * ys[b] (1024x768) -> f32 (d_out)
  wmma_gemm_kernel<64, 32, true, 0><<<dim3(cD / 32, cM / 64, cB), 256, 0, stream>>>(
      comb_bf, ys_bf, out, nullptr, nullptr, cD, cNP,
      (long)cM * cNP, (long)cNP * cD, cB, (long)cM * cD, 1);
}
